// Block_2791728742908
// MI455X (gfx1250) — matrix operand order repaired and checked by reading, not compiled, not run on hardware
//
#include <hip/hip_runtime.h>
#include <hip/hip_bf16.h>

typedef __bf16 bf16_t;
typedef __attribute__((ext_vector_type(16))) __bf16 v16bf;
typedef __attribute__((ext_vector_type(8)))  __bf16 v8bf;
typedef __attribute__((ext_vector_type(8)))  float  v8f;
typedef __attribute__((ext_vector_type(4)))  int    v4i;

#define DEV __device__ __forceinline__

#if __has_builtin(__builtin_amdgcn_global_load_async_to_lds_b128)
#define ASYNC_STAGE 1
#else
#define ASYNC_STAGE 0
#endif

// Copy 16B global -> LDS. Async (ASYNCcnt-tracked) when available.
DEV void stage16(const bf16_t* g, bf16_t* l) {
#if ASYNC_STAGE
  __builtin_amdgcn_global_load_async_to_lds_b128(
      (__attribute__((address_space(1))) v4i*)g,
      (__attribute__((address_space(3))) v4i*)l, 0, 0);
#else
  *(v8bf*)l = *(const v8bf*)g;
#endif
}

DEV void stage_wait() {
#if ASYNC_STAGE
#if __has_builtin(__builtin_amdgcn_s_wait_asynccnt)
  __builtin_amdgcn_s_wait_asynccnt(0);
#else
  asm volatile("s_wait_asynccnt 0" ::: "memory");
#endif
#endif
}

// ---------------------------------------------------------------------------
// WMMA fragment helpers (wave32, v_wmma_f32_16x16x32_bf16)
//
// A operand, 16x32 (MxK), row-major source, leading dim lda:
//   lanes 0-15 : row m=lane,   K = {k0..k0+7},   {k0+16..k0+23}
//   lanes 16-31: row m=lane-16,K = {k0+8..k0+15},{k0+24..k0+31}
DEV v16bf load_frag_a(const bf16_t* A, int lda, int row0, int k0, int lane) {
  const int m  = lane & 15;
  const int ko = (lane >> 4) << 3;               // 0 or 8
  const bf16_t* p = A + (row0 + m) * lda + k0 + ko;
  v16bf f;
  ((v8bf*)&f)[0] = *(const v8bf*)(p);
  ((v8bf*)&f)[1] = *(const v8bf*)(p + 16);
  return f;
}

// B operand, 32x16 (KxN), loaded from row-major B^T ("Bt"[N][K], ld=ldb):
//   lanes 0-15 : col n=lane, K = {k0..k0+7},{k0+16..k0+23} ; lanes 16-31: col n, K = {k0+8..k0+15},{k0+24..k0+31}
DEV v16bf load_frag_b(const bf16_t* Bt, int ldb, int col0, int k0, int lane) {
  const int n  = lane & 15;
  const int ko = (lane >> 4) << 3;               // 0 or 8
  const bf16_t* p = Bt + (col0 + n) * ldb + k0 + ko;
  v16bf f;
  ((v8bf*)&f)[0] = *(const v8bf*)(p);
  ((v8bf*)&f)[1] = *(const v8bf*)(p + 16);
  return f;
}

DEV v8f wmma_bf16(v16bf a, v16bf b, v8f c) {
  return __builtin_amdgcn_wmma_f32_16x16x32_bf16(
      false, a, false, b, (short)0, c, false, false);
}

// ---------------------------------------------------------------------------
// Weight fp32 [K][N]  ->  bf16 transposed [N][K]
__global__ __launch_bounds__(256) void transpose_to_bf16(
    const float* __restrict__ W, bf16_t* __restrict__ Wt, int K, int N) {
  int idx = blockIdx.x * 256 + threadIdx.x;
  if (idx < K * N) {
    int kk = idx / N, nn = idx - kk * N;
    Wt[nn * K + kk] = (bf16_t)W[idx];
  }
}

// ---------------------------------------------------------------------------
// LayerNorm over D=768, one block (256 threads) per row, bf16 output.
__global__ __launch_bounds__(256) void ln_bf16_kernel(
    const float* __restrict__ x, const float* __restrict__ g,
    const float* __restrict__ b, bf16_t* __restrict__ out) {
  const int row = blockIdx.x;
  const int tid = threadIdx.x;
  const float* xr = x + (size_t)row * 768;
  float v0 = xr[tid], v1 = xr[tid + 256], v2 = xr[tid + 512];
  float s  = v0 + v1 + v2;
  float sq = v0 * v0 + v1 * v1 + v2 * v2;
#pragma unroll
  for (int off = 1; off < 32; off <<= 1) {
    s  += __shfl_xor(s,  off, 32);
    sq += __shfl_xor(sq, off, 32);
  }
  __shared__ float sh1[8], sh2[8];
  const int wid = tid >> 5, lane = tid & 31;
  if (lane == 0) { sh1[wid] = s; sh2[wid] = sq; }
  __syncthreads();
  float ts = 0.f, tq = 0.f;
#pragma unroll
  for (int i = 0; i < 8; ++i) { ts += sh1[i]; tq += sh2[i]; }
  const float mu  = ts * (1.0f / 768.0f);
  const float var = tq * (1.0f / 768.0f) - mu * mu;
  const float rs  = rsqrtf(var + 1e-6f);
  bf16_t* orow = out + (size_t)row * 768;
  orow[tid]       = (bf16_t)((v0 - mu) * rs * g[tid]       + b[tid]);
  orow[tid + 256] = (bf16_t)((v1 - mu) * rs * g[tid + 256] + b[tid + 256]);
  orow[tid + 512] = (bf16_t)((v2 - mu) * rs * g[tid + 512] + b[tid + 512]);
}

// ---------------------------------------------------------------------------
// LDS-staged, double-buffered WMMA GEMM:
//   C[M,N] = A[M,K](bf16) x Bt[N,K]^T(bf16) + bias, + epilogue.
// Block: 256 threads = 8 waves; block tile 128x128; wave tile 64x32 (4x2 WMMA).
// Tiles staged via GLOBAL_LOAD_ASYNC_TO_LDS (ASYNCcnt) with bank-pad 40.
enum { EPI_Q = 0, EPI_K = 1, EPI_V = 2, EPI_ATTN = 3, EPI_GELU = 4, EPI_OUT = 5 };

#define APAD 40  // 32 payload + 8 pad bf16 -> 80B row stride, conflict-free

template <int EPI>
__global__ __launch_bounds__(256) void gemm_bf16_wmma(
    const bf16_t* __restrict__ A, const bf16_t* __restrict__ Bt,
    const float* __restrict__ bias, const float* __restrict__ resid,
    void* __restrict__ outp, int N, int K) {
  __shared__ __align__(16) bf16_t Abuf[2][128][APAD];
  __shared__ __align__(16) bf16_t Bbuf[2][128][APAD];
  const int tid  = threadIdx.x;
  const int lane = tid & 31;
  const int wid  = tid >> 5;
  const int rowBase = blockIdx.y * 128;
  const int colBase = blockIdx.x * 128;
  const int wrow = (wid & 1) * 64;   // wave tile inside block
  const int wcol = (wid >> 1) * 32;

  // staging: each thread copies 32B (16 bf16) of A and of B per K-step
  const int srow = tid >> 1;          // 0..127
  const int soff = (tid & 1) * 16;    // 0 or 16 (bf16 elems)
  const bf16_t* gA = A  + (size_t)(rowBase + srow) * K + soff;
  const bf16_t* gB = Bt + (size_t)(colBase + srow) * K + soff;

  auto stage = [&](int pb, int k0) {
    stage16(gA + k0,     &Abuf[pb][srow][soff]);
    stage16(gA + k0 + 8, &Abuf[pb][srow][soff + 8]);
    stage16(gB + k0,     &Bbuf[pb][srow][soff]);
    stage16(gB + k0 + 8, &Bbuf[pb][srow][soff + 8]);
  };

  stage(0, 0);
  v8f acc[4][2] = {};
  const int nk = K >> 5;
  for (int t = 0; t < nk; ++t) {
    const int p = t & 1;
    stage_wait();          // my async loads into buffer p are done
    __syncthreads();       // everyone's are done; prev compute on p^1 done
    if (t + 1 < nk) stage(p ^ 1, (t + 1) << 5);

    v16bf af[4];
#pragma unroll
    for (int i = 0; i < 4; ++i)
      af[i] = load_frag_a(&Abuf[p][0][0], APAD, wrow + i * 16, 0, lane);
    v16bf bfr[2];
#pragma unroll
    for (int j = 0; j < 2; ++j)
      bfr[j] = load_frag_b(&Bbuf[p][0][0], APAD, wcol + j * 16, 0, lane);
#pragma unroll
    for (int i = 0; i < 4; ++i)
#pragma unroll
      for (int j = 0; j < 2; ++j)
        acc[i][j] = wmma_bf16(af[i], bfr[j], acc[i][j]);
  }

  const int mhi = (lane >> 4) << 3;
#pragma unroll
  for (int i = 0; i < 4; ++i)
#pragma unroll
    for (int j = 0; j < 2; ++j)
#pragma unroll
      for (int r = 0; r < 8; ++r) {
        const int row = rowBase + wrow + i * 16 + mhi + r;
        const int col = colBase + wcol + j * 16 + (lane & 15);
        float v = acc[i][j][r] + bias[col];
        if constexpr (EPI == EPI_Q) {
          ((bf16_t*)outp)[row * N + col] = (bf16_t)(v * 0.125f);  // 1/sqrt(64)
        } else if constexpr (EPI == EPI_K) {
          ((bf16_t*)outp)[row * N + col] = (bf16_t)v;
        } else if constexpr (EPI == EPI_V) {
          // scatter to per-head V^T: [b][h][dh=64][tok=1024]
          const int bb = row >> 10, tok = row & 1023;
          const int hh = col >> 6,  dd  = col & 63;
          ((bf16_t*)outp)[(((bb * 12 + hh) << 6) + dd) * 1024 + tok] = (bf16_t)v;
        } else if constexpr (EPI == EPI_ATTN) {
          ((float*)outp)[row * N + col] = v + resid[row * N + col];
        } else if constexpr (EPI == EPI_GELU) {
          const float gl = 0.5f * v * (1.0f + erff(v * 0.70710678118654752f));
          ((bf16_t*)outp)[row * N + col] = (bf16_t)gl;
        } else {  // EPI_OUT
          ((float*)outp)[row * N + col] = v + resid[row * N + col];
        }
      }
}

// ---------------------------------------------------------------------------
// Flash attention, LDS-staged K/V chunks shared by all 8 waves of the block.
// One wave per 16-query tile; keys streamed in 32-key chunks, double-buffered.
// q pre-scaled by 1/sqrt(DH); vt is per-head V^T [b][h][64][1024].
#define KPAD 72  // 64 payload + 8 pad
#define VPAD 40  // 32 payload + 8 pad

__global__ __launch_bounds__(256) void attention_kernel(
    const bf16_t* __restrict__ q, const bf16_t* __restrict__ k,
    const bf16_t* __restrict__ vt, bf16_t* __restrict__ ctx) {
  __shared__ __align__(16) bf16_t Ktile[2][32][KPAD];  // [key][dh]
  __shared__ __align__(16) bf16_t Vtile[2][64][VPAD];  // [dh][key]
  __shared__ __align__(16) bf16_t Pbuf[8][16][VPAD];   // per-wave P (D->A relayout)
  const int tid  = threadIdx.x;
  const int lane = tid & 31;
  const int wid  = tid >> 5;
  const int h = blockIdx.y, b = blockIdx.z;
  const int q0 = blockIdx.x * 128 + wid * 16;

  const bf16_t* Qb = q  + (b * 1024) * 768 + h * 64;    // [tok][768]
  const bf16_t* Kb = k  + (b * 1024) * 768 + h * 64;    // [tok][768]
  const bf16_t* Vb = vt + ((b * 12 + h) * 64) * 1024;   // [64][1024]

  // staging: K chunk 32x64 (16B/thread), V chunk 64x32 (16B/thread)
  const int skey = tid >> 3, skoff = (tid & 7) * 8;
  const int sd   = tid >> 2, svoff = (tid & 3) * 8;
  auto stage = [&](int pb, int kc) {
    stage16(Kb + (kc + skey) * 768 + skoff, &Ktile[pb][skey][skoff]);
    stage16(Vb + sd * 1024 + kc + svoff,    &Vtile[pb][sd][svoff]);
  };

  // Q fragments are loop-invariant (DH = 64 => two K-steps)
  const v16bf aq0 = load_frag_a(Qb, 768, q0, 0,  lane);
  const v16bf aq1 = load_frag_a(Qb, 768, q0, 32, lane);

  v8f acc[4] = {};          // ctx accumulators: 16q x 64dh
  float row_m[8], row_l[8];
#pragma unroll
  for (int r = 0; r < 8; ++r) { row_m[r] = -1e30f; row_l[r] = 0.f; }
  const int mhi = (lane >> 4) << 3;

  stage(0, 0);
  for (int t = 0; t < 32; ++t) {
    const int p = t & 1;
    stage_wait();
    __syncthreads();
    if (t + 1 < 32) stage(p ^ 1, (t + 1) * 32);

    // S = Q K^T (scaled via pre-scaled q); keys are local rows of Ktile
    v8f s0 = {}, s1 = {};
    s0 = wmma_bf16(aq0, load_frag_b(&Ktile[p][0][0], KPAD, 0,  0,  lane), s0);
    s0 = wmma_bf16(aq1, load_frag_b(&Ktile[p][0][0], KPAD, 0,  32, lane), s0);
    s1 = wmma_bf16(aq0, load_frag_b(&Ktile[p][0][0], KPAD, 16, 0,  lane), s1);
    s1 = wmma_bf16(aq1, load_frag_b(&Ktile[p][0][0], KPAD, 16, 32, lane), s1);

    // Online softmax over this 32-key chunk. Each D-layout row lives in one
    // half-wave: xor masks 1,2,4,8 reduce across the row's 16 lanes.
#pragma unroll
    for (int r = 0; r < 8; ++r) {
      const float a0 = s0[r], a1 = s1[r];
      float vmax = fmaxf(a0, a1);
#pragma unroll
      for (int off = 1; off < 16; off <<= 1)
        vmax = fmaxf(vmax, __shfl_xor(vmax, off, 32));
      const float nm    = fmaxf(row_m[r], vmax);
      const float alpha = __expf(row_m[r] - nm);
      const float p0 = __expf(a0 - nm), p1 = __expf(a1 - nm);
      float ps = p0 + p1;
#pragma unroll
      for (int off = 1; off < 16; off <<= 1)
        ps += __shfl_xor(ps, off, 32);
      row_l[r] = row_l[r] * alpha + ps;
      row_m[r] = nm;
#pragma unroll
      for (int tt = 0; tt < 4; ++tt) acc[tt][r] *= alpha;
      const int m = mhi + r;
      Pbuf[wid][m][lane & 15]        = (bf16_t)p0;   // local keys 0..15
      Pbuf[wid][m][(lane & 15) + 16] = (bf16_t)p1;   // local keys 16..31
    }

    // ctx += P (16x32) x V (32x64); V^T rows contiguous in Vtile.
    const v16bf ap = load_frag_a(&Pbuf[wid][0][0], VPAD, 0, 0, lane);
#pragma unroll
    for (int tt = 0; tt < 4; ++tt)
      acc[tt] = wmma_bf16(ap, load_frag_b(&Vtile[p][0][0], VPAD, tt * 16, 0, lane),
                          acc[tt]);
  }

#pragma unroll
  for (int r = 0; r < 8; ++r) {
    const float inv = 1.0f / row_l[r];
    const int row = b * 1024 + q0 + mhi + r;
#pragma unroll
    for (int tt = 0; tt < 4; ++tt) {
      const int col = h * 64 + tt * 16 + (lane & 15);
      ctx[row * 768 + col] = (bf16_t)(acc[tt][r] * inv);
    }
  }
}

// ---------------------------------------------------------------------------
extern "C" void kernel_launch(void* const* d_in, const int* in_sizes, int n_in,
                              void* d_out, int out_size, void* d_ws, size_t ws_size,
                              hipStream_t stream) {
  (void)in_sizes; (void)n_in; (void)out_size; (void)ws_size;
  const float* x    = (const float*)d_in[0];
  const float* ln1g = (const float*)d_in[1];
  const float* ln1b = (const float*)d_in[2];
  const float* Wq   = (const float*)d_in[3];
  const float* bq   = (const float*)d_in[4];
  const float* Wk   = (const float*)d_in[5];
  const float* bk   = (const float*)d_in[6];
  const float* Wv   = (const float*)d_in[7];
  const float* bv   = (const float*)d_in[8];
  const float* Wo   = (const float*)d_in[9];
  const float* bo   = (const float*)d_in[10];
  const float* ln2g = (const float*)d_in[11];
  const float* ln2b = (const float*)d_in[12];
  const float* W1   = (const float*)d_in[13];
  const float* b1   = (const float*)d_in[14];
  const float* W2   = (const float*)d_in[15];
  const float* b2   = (const float*)d_in[16];

  const int M = 8192, D = 768, FF = 3072;

  char* wp = (char*)d_ws;
  auto alloc = [&](size_t bytes) -> void* {
    void* p = (void*)wp;
    wp += (bytes + 255) & ~(size_t)255;
    return p;
  };
  bf16_t* xn1  = (bf16_t*)alloc((size_t)M * D  * 2);
  bf16_t* qb   = (bf16_t*)alloc((size_t)M * D  * 2);
  bf16_t* kb   = (bf16_t*)alloc((size_t)M * D  * 2);
  bf16_t* vtb  = (bf16_t*)alloc((size_t)M * D  * 2);
  bf16_t* ctxb = (bf16_t*)alloc((size_t)M * D  * 2);
  float*  x2   = (float*) alloc((size_t)M * D  * 4);
  bf16_t* xn2  = (bf16_t*)alloc((size_t)M * D  * 2);
  bf16_t* h1   = (bf16_t*)alloc((size_t)M * FF * 2);
  bf16_t* wqT  = (bf16_t*)alloc((size_t)D * D  * 2);
  bf16_t* wkT  = (bf16_t*)alloc((size_t)D * D  * 2);
  bf16_t* wvT  = (bf16_t*)alloc((size_t)D * D  * 2);
  bf16_t* woT  = (bf16_t*)alloc((size_t)D * D  * 2);
  bf16_t* w1T  = (bf16_t*)alloc((size_t)D * FF * 2);
  bf16_t* w2T  = (bf16_t*)alloc((size_t)FF * D * 2);

  // Weight convert + transpose (bf16, B^T layout for WMMA B fragments)
  transpose_to_bf16<<<(D * D  + 255) / 256, 256, 0, stream>>>(Wq, wqT, D, D);
  transpose_to_bf16<<<(D * D  + 255) / 256, 256, 0, stream>>>(Wk, wkT, D, D);
  transpose_to_bf16<<<(D * D  + 255) / 256, 256, 0, stream>>>(Wv, wvT, D, D);
  transpose_to_bf16<<<(D * D  + 255) / 256, 256, 0, stream>>>(Wo, woT, D, D);
  transpose_to_bf16<<<(D * FF + 255) / 256, 256, 0, stream>>>(W1, w1T, D, FF);
  transpose_to_bf16<<<(FF * D + 255) / 256, 256, 0, stream>>>(W2, w2T, FF, D);

  const dim3 gD(D / 128, M / 128);    // (6, 64)
  const dim3 gF(FF / 128, M / 128);   // (24, 64)

  ln_bf16_kernel<<<M, 256, 0, stream>>>(x, ln1g, ln1b, xn1);
  gemm_bf16_wmma<EPI_Q>   <<<gD, 256, 0, stream>>>(xn1,  wqT, bq, nullptr, qb,  D,  D);
  gemm_bf16_wmma<EPI_K>   <<<gD, 256, 0, stream>>>(xn1,  wkT, bk, nullptr, kb,  D,  D);
  gemm_bf16_wmma<EPI_V>   <<<gD, 256, 0, stream>>>(xn1,  wvT, bv, nullptr, vtb, D,  D);
  attention_kernel<<<dim3(8, 12, 8), 256, 0, stream>>>(qb, kb, vtb, ctxb);
  gemm_bf16_wmma<EPI_ATTN><<<gD, 256, 0, stream>>>(ctxb, woT, bo, x,       x2,  D,  D);
  ln_bf16_kernel<<<M, 256, 0, stream>>>(x2, ln2g, ln2b, xn2);
  gemm_bf16_wmma<EPI_GELU><<<gF, 256, 0, stream>>>(xn2,  w1T, b1, nullptr, h1,  FF, D);
  gemm_bf16_wmma<EPI_OUT> <<<gD, 256, 0, stream>>>(h1,   w2T, b2, x2, (float*)d_out, D, FF);
}
